// MultiHeadAtt2d_71708773974967
// MI455X (gfx1250) — matrix operand order repaired and checked by reading, not compiled, not run on hardware
//
#include <hip/hip_runtime.h>
#include <math.h>

// ---------------------------------------------------------------------------
// MultiHeadAtt2d for MI455X (gfx1250): bf16 WMMA pipeline.
//   prep (input pad/convert, weight transpose/convert)
//   -> conv1d(q/k/v) [dual-layout outputs] -> time flash-attention
//   -> freq attention -> final FC
// All GEMM work uses v_wmma_f32_16x16x32_bf16 (f32 accumulate).
// Fragment loads are b128; GEMM loops are double-buffered for in-wave overlap.
// ---------------------------------------------------------------------------

typedef __bf16 bf16;
typedef __attribute__((ext_vector_type(16))) __bf16 v16bf;
typedef __attribute__((ext_vector_type(8)))  __bf16 v8bf;
typedef __attribute__((ext_vector_type(8)))  float  v8f;
typedef __attribute__((ext_vector_type(4)))  float  v4f;

#define Bb   8
#define Mm   1024
#define Dd   512
#define Cc   512
#define Hh   8
#define DKk  64
#define MP2  (Mm + 2)                       // padded rows per batch
#define SC        (1.0f / 512.0f)           // softmax(att / C), C == 512
#define MASKEDV   (-1.0e15f * (1.0f/512.0f))

union bfu { v16bf v; v8bf h[2]; };

// ---- WMMA helper ----------------------------------------------------------

__device__ __forceinline__ v8f wmma_bf16(v16bf a, v16bf b, v8f c) {
  return __builtin_amdgcn_wmma_f32_16x16x32_bf16(
      /*neg_a=*/false, a, /*neg_b=*/false, b,
      /*c_mod=*/(short)0, c, /*reuse_a=*/false, /*reuse_b=*/false);
}

// ---- Fragment loaders -----------------------------------------------------
// A fragment (16x32): lane l: row = l&15, hf = l>>4.
//   a[i] = A[row, hf*8+i], a[8+i] = A[row, 16+hf*8+i]
// B fragment (32x16): lane l: col = l&15, hf = l>>4.
//   b[i] = B[hf*8+i, col], b[8+i] = B[16+hf*8+i, col]

// A from row-major [row][k] storage (contiguous k): two b128 loads.
__device__ __forceinline__ v16bf load_a_rowmajor(const bf16* p, int ld) {
  const int l = threadIdx.x & 31, row = l & 15, hf = l >> 4;
  const bf16* q0 = p + (size_t)row * ld + hf * 8;
  bfu u;
  u.h[0] = *(const v8bf*)(q0);
  u.h[1] = *(const v8bf*)(q0 + 16);
  return u.v;
}

// A from [k][row] storage (stride ld between k): scalar, lane-coalesced.
// Only used once per block (time-attn Q tile).
__device__ __forceinline__ v16bf load_a_kmajor(const bf16* p, int ld) {
  const int l = threadIdx.x & 31, row = l & 15, hf = l >> 4;
  v16bf a;
#pragma unroll
  for (int i = 0; i < 8; ++i) {
    a[i]     = p[(size_t)(hf * 8 + i) * ld + row];
    a[8 + i] = p[(size_t)(16 + hf * 8 + i) * ld + row];
  }
  return a;
}

// B from [n][k] storage (contiguous k per lane): two b128 loads.
__device__ __forceinline__ v16bf load_b_nmajor(const bf16* p, int ld) {
  const int l = threadIdx.x & 31, col = l & 15, hf = l >> 4;
  const bf16* q0 = p + (size_t)col * ld + hf * 8;
  bfu u;
  u.h[0] = *(const v8bf*)(q0);
  u.h[1] = *(const v8bf*)(q0 + 16);
  return u.v;
}

// Row-wise reductions across the 16-lane halves of the C/D layout.
__device__ __forceinline__ float half16_max(float x) {
#pragma unroll
  for (int m = 1; m <= 8; m <<= 1) x = fmaxf(x, __shfl_xor(x, m, 32));
  return x;
}
__device__ __forceinline__ float half16_sum(float x) {
#pragma unroll
  for (int m = 1; m <= 8; m <<= 1) x += __shfl_xor(x, m, 32);
  return x;
}

// ---- Prep kernels ---------------------------------------------------------

// x[B][M][D] fp32 -> xpad[B][M+2][D] bf16 with zero rows at mm==0 and mm==M+1.
__global__ void input_pad_bf16(const float* __restrict__ in, bf16* __restrict__ out) {
  const size_t idx = ((size_t)blockIdx.x * 256 + threadIdx.x) * 8;
  if (idx >= (size_t)Bb * MP2 * Dd) return;
  const int d = (int)(idx % Dd);
  const size_t prow = idx / Dd;
  const int mm = (int)(prow % MP2);
  const int b  = (int)(prow / MP2);
  v8bf o;
  if (mm == 0 || mm == Mm + 1) {
#pragma unroll
    for (int i = 0; i < 8; ++i) o[i] = (bf16)0.0f;
  } else {
    const float* p = in + ((size_t)b * Mm + (mm - 1)) * Dd + d;
    v4f f0 = *(const v4f*)(p);
    v4f f1 = *(const v4f*)(p + 4);
#pragma unroll
    for (int i = 0; i < 4; ++i) { o[i] = (bf16)f0[i]; o[4 + i] = (bf16)f1[i]; }
  }
  *(v8bf*)(out + idx) = o;
}

// w[C][D][3] fp32 -> wT[3][C][D] bf16  (B fragments become contiguous b128)
__global__ void conv_w_prep(const float* __restrict__ w, bf16* __restrict__ wT) {
  const int idx = blockIdx.x * 256 + threadIdx.x;   // over C*D
  if (idx >= Cc * Dd) return;
  const int c = idx / Dd, d = idx % Dd;
#pragma unroll
  for (int t = 0; t < 3; ++t) {
    wT[((size_t)t * Cc + c) * Dd + d] = (bf16)w[((size_t)c * Dd + d) * 3 + t];
  }
}

// fp32 -> bf16, 8 elements per thread.
__global__ void f32_to_bf16_vec8(const float* __restrict__ in,
                                 bf16* __restrict__ out, int n) {
  const int idx = (blockIdx.x * 256 + threadIdx.x) * 8;
  if (idx >= n) return;
  v4f f0 = *(const v4f*)(in + idx);
  v4f f1 = *(const v4f*)(in + idx + 4);
  v8bf o;
#pragma unroll
  for (int i = 0; i < 4; ++i) { o[i] = (bf16)f0[i]; o[4 + i] = (bf16)f1[i]; }
  *(v8bf*)(out + idx) = o;
}

// ---- Kernel 1: conv1d "same" projection as 3 shifted WMMA GEMMs -----------
// out[b,m,c] = (sum_{t} sum_d xpad[b, m+t, d] * wT[t, c, d] + bias[c]) * scale
// One wave computes a 32(m) x 64(c) tile; loads are double-buffered so the
// next chunk's fragments are in flight during the current 8 WMMAs.
__global__ void conv_proj_kernel(const bf16* __restrict__ xpad,  // [B][M+2][D]
                                 const bf16* __restrict__ wT,    // [3][C][D]
                                 const float* __restrict__ bias,
                                 float scale,
                                 bf16* __restrict__ outN,        // [B*M][C] | null
                                 bf16* __restrict__ outT) {      // [B*H*DK][M] | null
  const int strip  = blockIdx.x;       // (B*M/32) * (C/64)
  const int nstrip = strip & 7;        // C/64 = 8
  const int mtile  = strip >> 3;
  const int m0 = mtile * 32;
  const int n0 = nstrip * 64;
  const int b  = m0 / Mm;              // 1024 % 32 == 0: never crosses batch
  const int mm0 = m0 % Mm;

  v8f acc[2][4] = {};
  for (int t = 0; t < 3; ++t) {
    const bf16* ap0 = xpad + ((size_t)b * MP2 + mm0 + t) * Dd;   // rows m-1+t
    const bf16* ap1 = ap0 + (size_t)16 * Dd;
    const bf16* wp  = wT + ((size_t)t * Cc + n0) * Dd;

    v16bf a0 = load_a_rowmajor(ap0, Dd);
    v16bf a1 = load_a_rowmajor(ap1, Dd);
    v16bf bw[4];
#pragma unroll
    for (int nt = 0; nt < 4; ++nt) bw[nt] = load_b_nmajor(wp + (size_t)nt * 16 * Dd, Dd);

    for (int kk = 0; kk < Dd; kk += 32) {
      v16bf a0n, a1n, bwn[4];
      const int kn = kk + 32;
      if (kn < Dd) {                    // prefetch next chunk before consuming
        a0n = load_a_rowmajor(ap0 + kn, Dd);
        a1n = load_a_rowmajor(ap1 + kn, Dd);
#pragma unroll
        for (int nt = 0; nt < 4; ++nt)
          bwn[nt] = load_b_nmajor(wp + (size_t)nt * 16 * Dd + kn, Dd);
      }
#pragma unroll
      for (int nt = 0; nt < 4; ++nt) {
        acc[0][nt] = wmma_bf16(a0, bw[nt], acc[0][nt]);
        acc[1][nt] = wmma_bf16(a1, bw[nt], acc[1][nt]);
      }
      a0 = a0n; a1 = a1n;
#pragma unroll
      for (int nt = 0; nt < 4; ++nt) bw[nt] = bwn[nt];
    }
  }

  const int l = threadIdx.x & 31;
  const int col = l & 15, hf = l >> 4;
#pragma unroll
  for (int mi = 0; mi < 2; ++mi) {
#pragma unroll
    for (int nt = 0; nt < 4; ++nt) {
      const int c = n0 + nt * 16 + col;
      const float bv = bias[c];
      float vals[8];
#pragma unroll
      for (int r = 0; r < 8; ++r) vals[r] = (acc[mi][nt][r] + bv) * scale;
      if (outN) {
#pragma unroll
        for (int r = 0; r < 8; ++r) {
          outN[(size_t)(m0 + mi * 16 + r + 8 * hf) * Cc + c] = (bf16)vals[r];
        }
      }
      if (outT) {
        // lane's 8 rows are m = m0 + mi*16 + 8*hf + r: one b128 store
        v8bf pack;
#pragma unroll
        for (int r = 0; r < 8; ++r) pack[r] = (bf16)vals[r];
        const int h = c >> 6, d = c & 63;
        *(v8bf*)(outT + (((size_t)b * Hh + h) * DKk + d) * Mm
                 + m0 + mi * 16 + 8 * hf) = pack;
      }
    }
  }
}

// ---- Kernel 2: time attention (flash style) -------------------------------
// Per block (1 wave): one (b, h, 16 query rows). Streams keys in chunks of 32
// with running max/sum; P staged through LDS to convert C-layout -> A-layout.
// K fragment loads issue first, V fragment loads second: score WMMAs only wait
// for the K portion, so V stays in flight through the whole softmax section.
__global__ void time_attn_kernel(const bf16* __restrict__ qT,    // [B*H*DK][M]
                                 const bf16* __restrict__ kmat,  // [B*M][C]
                                 const bf16* __restrict__ vT,    // [B*H*DK][M]
                                 const unsigned char* __restrict__ mask,
                                 bf16* __restrict__ t_out) {     // [B*M][C]
  __shared__ bf16 plds[16 * 32];

  const int bid = blockIdx.x;          // B * H * (M/16)
  const int qtile = bid & 63;
  const int h = (bid >> 6) & 7;
  const int b = bid >> 9;
  const int m0 = qtile * 16;
  const size_t cbase = (size_t)b * Mm * Cc + (size_t)h * DKk;    // [m][c] layouts
  const size_t tbase = ((size_t)b * Hh + h) * DKk * Mm;          // [d][m] layouts

  const int l = threadIdx.x & 31;
  const int col = l & 15, hf = l >> 4;

  // Q tile (16 x 64) as two A fragments from qT[d][m] (loaded once).
  v16bf qa[2];
#pragma unroll
  for (int c = 0; c < 2; ++c) {
    qa[c] = load_a_kmajor(qT + tbase + (size_t)c * 32 * Mm + m0, Mm);
  }

  bool mq[8];
#pragma unroll
  for (int r = 0; r < 8; ++r) mq[r] = mask[b * Mm + m0 + r + 8 * hf] != 0;

  float rmax[8], rsum[8];
  v8f o[4] = {};
#pragma unroll
  for (int r = 0; r < 8; ++r) { rmax[r] = -INFINITY; rsum[r] = 0.0f; }

  for (int kb = 0; kb < Mm; kb += 32) {
    // K fragments first (score WMMAs wait only on these)...
    v16bf bk[2][2];
#pragma unroll
    for (int kt = 0; kt < 2; ++kt)
#pragma unroll
      for (int c = 0; c < 2; ++c)
        bk[kt][c] = load_b_nmajor(kmat + cbase + (size_t)(kb + kt * 16) * Cc + c * 32, Cc);
    // ...then V fragments (consumed only after softmax).
    v16bf bv[4];
#pragma unroll
    for (int nt = 0; nt < 4; ++nt)
      bv[nt] = load_b_nmajor(vT + tbase + (size_t)(nt * 16) * Mm + kb, Mm);

    v8f s[2] = {};
#pragma unroll
    for (int kt = 0; kt < 2; ++kt)
#pragma unroll
      for (int c = 0; c < 2; ++c)
        s[kt] = wmma_bf16(qa[c], bk[kt][c], s[kt]);

    const bool mk0 = mask[b * Mm + kb + col] != 0;
    const bool mk1 = mask[b * Mm + kb + 16 + col] != 0;

#pragma unroll
    for (int r = 0; r < 8; ++r) {
      float s0 = (mq[r] && mk0) ? s[0][r] : MASKEDV;   // q pre-scaled by 1/512
      float s1 = (mq[r] && mk1) ? s[1][r] : MASKEDV;
      float bmax = half16_max(fmaxf(s0, s1));
      float nm = fmaxf(rmax[r], bmax);
      float corr = __expf(rmax[r] - nm);               // 0 on first block
      float p0 = __expf(s0 - nm);
      float p1 = __expf(s1 - nm);
      rsum[r] = rsum[r] * corr + half16_sum(p0 + p1);
      rmax[r] = nm;
      const int prow = r + 8 * hf;
      plds[prow * 32 + col]      = (bf16)p0;
      plds[prow * 32 + 16 + col] = (bf16)p1;
#pragma unroll
      for (int nt = 0; nt < 4; ++nt) o[nt][r] *= corr;
    }
    __syncthreads();

    v16bf pa = load_a_rowmajor(plds, 32);
    __syncthreads();

#pragma unroll
    for (int nt = 0; nt < 4; ++nt) o[nt] = wmma_bf16(pa, bv[nt], o[nt]);
  }

#pragma unroll
  for (int r = 0; r < 8; ++r) {
    const float inv = 1.0f / rsum[r];
    const int orow = r + 8 * hf;
#pragma unroll
    for (int nt = 0; nt < 4; ++nt) {
      t_out[cbase + (size_t)(m0 + orow) * Cc + nt * 16 + col] = (bf16)(o[nt][r] * inv);
    }
  }
}

// ---- Kernel 3: frequency attention ----------------------------------------
// Block = 4 waves, one (b,h). Wave w owns rows i in [16w,16w+16) of the 64x64
// score matrix (K over all M=1024), wave-local softmax, then P(16x64) x V
// swept over all M, scattered to f_out[b, m, i*H + h]. Both loops are
// double-buffered.
__global__ void freq_attn_kernel(const bf16* __restrict__ qT,    // [B*H*DK][M]
                                 const bf16* __restrict__ kT,    // [B*H*DK][M]
                                 const bf16* __restrict__ v,     // [B*M][C]
                                 bf16* __restrict__ f_out) {     // [B*M][C]
  __shared__ bf16 plds[4][16 * 64];

  const int wave = threadIdx.x >> 5;
  const int bh = blockIdx.x;           // b*H + h
  const int b = bh >> 3;
  const int h = bh & 7;
  const size_t tbase = (size_t)bh * DKk * Mm;
  const size_t cbase = (size_t)b * Mm * Cc + (size_t)h * DKk;
  const int i0 = wave * 16;

  const int l = threadIdx.x & 31;
  const int col = l & 15, hf = l >> 4;

  v8f s[4] = {};
  {
    v16bf a = load_a_rowmajor(qT + tbase + (size_t)i0 * Mm, Mm);
    v16bf bk[4];
#pragma unroll
    for (int jt = 0; jt < 4; ++jt)
      bk[jt] = load_b_nmajor(kT + tbase + (size_t)(jt * 16) * Mm, Mm);

    for (int kk = 0; kk < Mm; kk += 32) {
      v16bf an, bkn[4];
      const int kn = kk + 32;
      if (kn < Mm) {
        an = load_a_rowmajor(qT + tbase + (size_t)i0 * Mm + kn, Mm);
#pragma unroll
        for (int jt = 0; jt < 4; ++jt)
          bkn[jt] = load_b_nmajor(kT + tbase + (size_t)(jt * 16) * Mm + kn, Mm);
      }
#pragma unroll
      for (int jt = 0; jt < 4; ++jt) s[jt] = wmma_bf16(a, bk[jt], s[jt]);
      a = an;
#pragma unroll
      for (int jt = 0; jt < 4; ++jt) bk[jt] = bkn[jt];
    }
  }

  // Softmax over j (64 cols: 4 tiles x 16 lanes); q pre-scaled by 1/512.
#pragma unroll
  for (int r = 0; r < 8; ++r) {
    float x0 = s[0][r], x1 = s[1][r], x2 = s[2][r], x3 = s[3][r];
    float mx = half16_max(fmaxf(fmaxf(x0, x1), fmaxf(x2, x3)));
    float e0 = __expf(x0 - mx), e1 = __expf(x1 - mx);
    float e2 = __expf(x2 - mx), e3 = __expf(x3 - mx);
    float inv = 1.0f / half16_sum(e0 + e1 + e2 + e3);
    const int prow = r + 8 * hf;
    plds[wave][prow * 64 + 0 * 16 + col] = (bf16)(e0 * inv);
    plds[wave][prow * 64 + 1 * 16 + col] = (bf16)(e1 * inv);
    plds[wave][prow * 64 + 2 * 16 + col] = (bf16)(e2 * inv);
    plds[wave][prow * 64 + 3 * 16 + col] = (bf16)(e3 * inv);
  }
  __syncthreads();

  v16bf pa[2];
#pragma unroll
  for (int c = 0; c < 2; ++c) {
    pa[c] = load_a_rowmajor(plds[wave] + c * 32, 64);
  }

  // P x V sweep over all m tiles, double-buffered.
  v16bf bv[2];
#pragma unroll
  for (int c = 0; c < 2; ++c)
    bv[c] = load_b_nmajor(v + cbase + c * 32, Cc);

  for (int mt = 0; mt < Mm / 16; ++mt) {
    v16bf bvn[2];
    if (mt + 1 < Mm / 16) {
#pragma unroll
      for (int c = 0; c < 2; ++c)
        bvn[c] = load_b_nmajor(v + cbase + (size_t)((mt + 1) * 16) * Cc + c * 32, Cc);
    }
    v8f o = {};
#pragma unroll
    for (int c = 0; c < 2; ++c) o = wmma_bf16(pa[c], bv[c], o);
    // f_out channel order is i*H + h (transpose(0,3,2,1) in the reference).
#pragma unroll
    for (int r = 0; r < 8; ++r) {
      const int i = i0 + r + 8 * hf;           // row of C tile = DK index
      const int m = mt * 16 + col;             // col of C tile = time index
      f_out[(size_t)(b * Mm + m) * Cc + i * Hh + h] = (bf16)o[r];
    }
    bv[0] = bvn[0]; bv[1] = bvn[1];
  }
}

// ---- Kernel 4: final FC ---------------------------------------------------
// out[bm, d] = sum_c t_out[bm,c]*fcW[d,c] + sum_c f_out[bm,c]*fcW[d,512+c] + fcb[d]
// One wave computes a 32 x 64 tile; double-buffered loads.
__global__ void fc_kernel(const bf16* __restrict__ t_out,
                          const bf16* __restrict__ f_out,
                          const bf16* __restrict__ fcWb,   // [D][2C] bf16
                          const float* __restrict__ fcb,
                          float* __restrict__ out) {
  const int strip  = blockIdx.x;       // (B*M/32) * (D/64)
  const int nstrip = strip & 7;        // D/64 = 8
  const int mtile  = strip >> 3;
  const int m0 = mtile * 32;
  const int n0 = nstrip * 64;

  v8f acc[2][4] = {};
#pragma unroll
  for (int hsel = 0; hsel < 2; ++hsel) {
    const bf16* src = hsel ? f_out : t_out;
    const bf16* wp  = fcWb + (size_t)n0 * (2 * Cc) + hsel * Cc;

    v16bf a0 = load_a_rowmajor(src + (size_t)m0 * Cc, Cc);
    v16bf a1 = load_a_rowmajor(src + (size_t)(m0 + 16) * Cc, Cc);
    v16bf bw[4];
#pragma unroll
    for (int nt = 0; nt < 4; ++nt)
      bw[nt] = load_b_nmajor(wp + (size_t)nt * 16 * (2 * Cc), 2 * Cc);

    for (int kk = 0; kk < Cc; kk += 32) {
      v16bf a0n, a1n, bwn[4];
      const int kn = kk + 32;
      if (kn < Cc) {
        a0n = load_a_rowmajor(src + (size_t)m0 * Cc + kn, Cc);
        a1n = load_a_rowmajor(src + (size_t)(m0 + 16) * Cc + kn, Cc);
#pragma unroll
        for (int nt = 0; nt < 4; ++nt)
          bwn[nt] = load_b_nmajor(wp + (size_t)nt * 16 * (2 * Cc) + kn, 2 * Cc);
      }
#pragma unroll
      for (int nt = 0; nt < 4; ++nt) {
        acc[0][nt] = wmma_bf16(a0, bw[nt], acc[0][nt]);
        acc[1][nt] = wmma_bf16(a1, bw[nt], acc[1][nt]);
      }
      a0 = a0n; a1 = a1n;
#pragma unroll
      for (int nt = 0; nt < 4; ++nt) bw[nt] = bwn[nt];
    }
  }

  const int l = threadIdx.x & 31;
  const int col = l & 15, hf = l >> 4;
#pragma unroll
  for (int mi = 0; mi < 2; ++mi) {
#pragma unroll
    for (int nt = 0; nt < 4; ++nt) {
      const float bv = fcb[n0 + nt * 16 + col];
#pragma unroll
      for (int r = 0; r < 8; ++r) {
        out[(size_t)(m0 + mi * 16 + r + 8 * hf) * Dd + n0 + nt * 16 + col] =
            acc[mi][nt][r] + bv;
      }
    }
  }
}

// ---- Host-side launch -----------------------------------------------------

extern "C" void kernel_launch(void* const* d_in, const int* in_sizes, int n_in,
                              void* d_out, int out_size, void* d_ws, size_t ws_size,
                              hipStream_t stream) {
  (void)in_sizes; (void)n_in; (void)out_size; (void)ws_size;

  const float* key   = (const float*)d_in[0];
  const float* query = (const float*)d_in[1];
  const float* value = (const float*)d_in[2];
  const unsigned char* mask = (const unsigned char*)d_in[3];
  const float* qW  = (const float*)d_in[4];
  const float* qb  = (const float*)d_in[5];
  const float* kW  = (const float*)d_in[6];
  const float* kb  = (const float*)d_in[7];
  const float* vW  = (const float*)d_in[8];
  const float* vb  = (const float*)d_in[9];
  const float* fcW = (const float*)d_in[10];
  const float* fcb = (const float*)d_in[11];

  const size_t PADSZ = (size_t)Bb * MP2 * Dd;   // padded bf16 input
  const size_t SZ    = (size_t)Bb * Mm * Cc;    // projection buffers
  const size_t WTS   = (size_t)3 * Cc * Dd;     // conv weights (transposed)
  const size_t FCWN  = (size_t)Dd * 2 * Cc;     // fc weights

  bf16* p = (bf16*)d_ws;
  bf16* xqp   = p; p += PADSZ;
  bf16* xkp   = p; p += PADSZ;
  bf16* xvp   = p; p += PADSZ;
  bf16* qT    = p; p += SZ;
  bf16* kT    = p; p += SZ;
  bf16* vT    = p; p += SZ;
  bf16* kproj = p; p += SZ;
  bf16* vproj = p; p += SZ;
  bf16* tbuf  = p; p += SZ;
  bf16* fbuf  = p; p += SZ;
  bf16* qwT   = p; p += WTS;
  bf16* kwT   = p; p += WTS;
  bf16* vwT   = p; p += WTS;
  bf16* fcWb  = p; p += FCWN;                   // total ~86 MB

  // Input conversion (padded) + weight prep.
  const int pad_blocks = (int)((PADSZ / 8 + 255) / 256);
  input_pad_bf16<<<pad_blocks, 256, 0, stream>>>(query, xqp);
  input_pad_bf16<<<pad_blocks, 256, 0, stream>>>(key,   xkp);
  input_pad_bf16<<<pad_blocks, 256, 0, stream>>>(value, xvp);
  const int cd_blocks = (Cc * Dd + 255) / 256;
  conv_w_prep<<<cd_blocks, 256, 0, stream>>>(qW, qwT);
  conv_w_prep<<<cd_blocks, 256, 0, stream>>>(kW, kwT);
  conv_w_prep<<<cd_blocks, 256, 0, stream>>>(vW, vwT);
  f32_to_bf16_vec8<<<(int)((FCWN / 8 + 255) / 256), 256, 0, stream>>>(fcW, fcWb, (int)FCWN);

  // Conv projections (32x64 tiles). q gets the 1/512 softmax scale folded in
  // and only needs the transposed layout; k/v need both layouts.
  const int conv_tiles = (Bb * Mm / 32) * (Cc / 64);    // 2048
  conv_proj_kernel<<<conv_tiles, 32, 0, stream>>>(xqp, qwT, qb, SC,   nullptr, qT);
  conv_proj_kernel<<<conv_tiles, 32, 0, stream>>>(xkp, kwT, kb, 1.0f, kproj,  kT);
  conv_proj_kernel<<<conv_tiles, 32, 0, stream>>>(xvp, vwT, vb, 1.0f, vproj,  vT);

  time_attn_kernel<<<Bb * Hh * (Mm / 16), 32, 0, stream>>>(qT, kproj, vT, mask, tbuf);
  freq_attn_kernel<<<Bb * Hh, 128, 0, stream>>>(qT, kT, vproj, fbuf);

  const int fc_tiles = (Bb * Mm / 32) * (Dd / 64);      // 2048
  fc_kernel<<<fc_tiles, 32, 0, stream>>>(tbuf, fbuf, fcWb, fcb, (float*)d_out);
}
